// LRGeneratorExp_10058813407438
// MI455X (gfx1250) — matrix operand order repaired and checked by reading, not compiled, not run on hardware
//
#include <hip/hip_runtime.h>
#include <hip/hip_bf16.h>

// ---------------- CDNA5 WMMA types/helpers ----------------
typedef __attribute__((ext_vector_type(16))) __bf16 v16bf;
typedef __attribute__((ext_vector_type(8)))  float  v8f;

__device__ __forceinline__ unsigned short f2bf(float f) {
    unsigned int u = __float_as_uint(f);
    unsigned int r = u + 0x7FFFu + ((u >> 16) & 1u);   // round-to-nearest-even
    return (unsigned short)(r >> 16);
}
__device__ __forceinline__ float bf2f(unsigned short b) {
    return __uint_as_float(((unsigned int)b) << 16);
}
__device__ __forceinline__ v8f vzero8() {
    v8f v;
#pragma unroll
    for (int i = 0; i < 8; i++) v[i] = 0.0f;
    return v;
}
__device__ __forceinline__ v8f wmma_bf(v16bf a, v16bf b, v8f c) {
    return __builtin_amdgcn_wmma_f32_16x16x32_bf16(false, a, false, b, (short)0, c, false, false);
}
// coalesced 32B fragment load from packed global (A-packed or B-packed)
__device__ __forceinline__ v16bf loadfrag(const unsigned short* p) {
    union { v16bf v; uint4 q[2]; } t;
    const uint4* g = (const uint4*)p;
    t.q[0] = g[0]; t.q[1] = g[1];
    return t.v;
}
// A-fragment gather from an LDS row (row-major), K(e) = kb + (e&7) + 16*(e>=8)
__device__ __forceinline__ v16bf gather_nl(const unsigned short* row, int kb) {
    union { v16bf v; unsigned short u[16]; } t;
#pragma unroll
    for (int e = 0; e < 16; e++) t.u[e] = row[kb + (e & 7) + ((e >> 3) << 4)];
    return t.v;
}
// same, with LayerNorm fused into the gather
__device__ __forceinline__ v16bf gather_ln(const unsigned short* row, int kb, float mu, float rs) {
    union { v16bf v; unsigned short u[16]; } t;
#pragma unroll
    for (int e = 0; e < 16; e++) {
        float x = bf2f(row[kb + (e & 7) + ((e >> 3) << 4)]);
        t.u[e] = f2bf((x - mu) * rs);
    }
    return t.v;
}
__device__ __forceinline__ float gelu(float x) {
    return 0.5f * x * (1.0f + erff(x * 0.70710678118654752f));
}

// ---------------- K0: pack fp32 weight [K][N] into B-fragment bf16 layout ----------------
// tile = nt*KS + ks ; within tile: [lane][e], value = W[ks*32 + (e&7) + 8*(lane>=16) + 16*(e>=8)][nt*16 + (lane&15)]
__global__ void k_pack(const float* __restrict__ W, unsigned short* __restrict__ dst, int K, int N) {
    int tid = blockIdx.x * blockDim.x + threadIdx.x;
    if (tid >= K * N) return;
    int e = tid & 15;
    int ln = (tid >> 4) & 31;
    int tile = tid >> 9;
    int KS = K >> 5;
    int ks = tile % KS;
    int nt = tile / KS;
    int k = ks * 32 + (e & 7) + ((ln >> 4) << 3) + ((e >> 3) << 4);
    int n = nt * 16 + (ln & 15);
    dst[tid] = f2bf(W[(size_t)k * N + n]);
}

__global__ void k_zero(float* __restrict__ p, int n) {
    int tid = blockIdx.x * blockDim.x + threadIdx.x;
    if (tid < n) p[tid] = 0.0f;
}

// ---------------- K1: depthwise 3x3 conv + posenc + patchify into A-packed bf16 ----------------
__global__ void k_conv(const float* __restrict__ x, const float* __restrict__ cw,
                       unsigned short* __restrict__ P) {
    int tid = blockIdx.x * blockDim.x + threadIdx.x;   // 8*64*128*128
    int ww = tid & 127;
    int hh = (tid >> 7) & 127;
    int c  = (tid >> 14) & 63;
    int b  = tid >> 20;
    const float* xc = x + (((size_t)b * 64 + c) << 14);
    float s = 0.0f;
#pragma unroll
    for (int kh = 0; kh < 3; kh++) {
#pragma unroll
        for (int kw = 0; kw < 3; kw++) {
            int h2 = hh + kh - 1, w2 = ww + kw - 1;
            if (h2 >= 0 && h2 < 128 && w2 >= 0 && w2 < 128)
                s += xc[h2 * 128 + w2] * cw[c * 9 + kh * 3 + kw];
        }
    }
    // positional encoding: c<32 uses h, c>=32 uses w; (even=sin, odd=cos), i=cc/2, q=16
    int cc2 = (c < 32) ? c : (c - 32);
    float coord = (float)((c < 32) ? hh : ww);
    int i = cc2 >> 1;
    float arg = coord * expf((float)i * -0.57564627324851147f);  // 10000^{-i/16}
    float pos = (cc2 & 1) ? cosf(arg) : sinf(arg);
    float v = s + pos;
    // scatter into A-fragment-packed patch matrix (KS=2)
    int pi = hh >> 3, r8 = hh & 7, pj = ww >> 3, s8 = ww & 7;
    int m = (((b * 64 + c) * 16) + pi) * 16 + pj;
    int k = r8 * 8 + s8;
    int tile = m >> 4, r = m & 15;
    int ks = k >> 5, kk = k & 31;
    int h = (kk >> 3) & 1;
    int e = (kk & 7) + ((kk >> 4) << 3);
    P[(size_t)(((tile * 2 + ks) * 32) + (h << 4) + r) * 16 + e] = f2bf(v);
}

// ---------------- K2: fused BTokenizer + pi/pj mean reduction ----------------
// grid: (b*c)*4 pig blocks, 128 threads (4 waves). wave w owns pi = pig*4+w, all 16 pj.
__global__ __launch_bounds__(128) void k_btok(
    const unsigned short* __restrict__ P,
    const unsigned short* __restrict__ binP,
    const unsigned short* __restrict__ blkP,
    const unsigned short* __restrict__ boutP,
    const float* __restrict__ bin_b,
    const float* __restrict__ blk_b,
    const float* __restrict__ bout_b,
    float* __restrict__ vmean,
    float* __restrict__ hmean) {
    __shared__ unsigned short tS[64 * 258];
    __shared__ float vpart[16 * 257];

    const int lane = threadIdx.x & 31;
    const int wv   = threadIdx.x >> 5;
    const int half = lane >> 4;
    const int cc   = lane & 15;
    const int pig  = blockIdx.x & 3;
    const int bc   = blockIdx.x >> 2;
    const int pi   = pig * 4 + wv;
    const int tile = bc * 16 + pi;          // token A-tile (16 pj rows)
    unsigned short* baseT = tS + wv * 16 * 258;

    // ---- bin: patches(64) -> 256 ----
    {
        v8f acc[16];
#pragma unroll
        for (int nt = 0; nt < 16; nt++) acc[nt] = vzero8();
#pragma unroll
        for (int ks = 0; ks < 2; ks++) {
            v16bf a = loadfrag(P + (size_t)((tile * 2 + ks) * 32 + lane) * 16);
#pragma unroll
            for (int nt = 0; nt < 16; nt++) {
                v16bf bb = loadfrag(binP + (size_t)(nt * 2 + ks) * 512 + lane * 16);
                acc[nt] = wmma_bf(a, bb, acc[nt]);
            }
        }
#pragma unroll
        for (int nt = 0; nt < 16; nt++) {
            int n = nt * 16 + cc;
            float bia = bin_b[n];
#pragma unroll
            for (int j = 0; j < 8; j++) {
                int rr = j + half * 8;
                baseT[rr * 258 + n] = f2bf(acc[nt][j] + bia);
            }
        }
    }

    const unsigned short* rowp = baseT + cc * 258;

    // ---- 4 residual blocks: t += gelu(LN(t) @ W + b) ----
    for (int blk = 0; blk < 4; blk++) {
        float s = 0.0f, sq = 0.0f;
        int n0 = half << 7;
        for (int n = n0; n < n0 + 128; n++) {
            float xv = bf2f(rowp[n]);
            s += xv; sq += xv * xv;
        }
        s  += __shfl_xor(s, 16);
        sq += __shfl_xor(sq, 16);
        float mu = s * (1.0f / 256.0f);
        float rs = rsqrtf(sq * (1.0f / 256.0f) - mu * mu + 1e-5f);

        const unsigned short* wb = blkP + (size_t)blk * 65536;
        const float* bbp = blk_b + blk * 256;
        v8f acc[16];
#pragma unroll
        for (int nt = 0; nt < 16; nt++) acc[nt] = vzero8();
#pragma unroll
        for (int ks = 0; ks < 8; ks++) {
            v16bf a = gather_ln(rowp, ks * 32 + half * 8, mu, rs);
#pragma unroll
            for (int nt = 0; nt < 16; nt++) {
                v16bf bb = loadfrag(wb + (size_t)(nt * 8 + ks) * 512 + lane * 16);
                acc[nt] = wmma_bf(a, bb, acc[nt]);
            }
        }
#pragma unroll
        for (int nt = 0; nt < 16; nt++) {
            int n = nt * 16 + cc;
            float bia = bbp[n];
#pragma unroll
            for (int j = 0; j < 8; j++) {
                int rr = j + half * 8;
                int idx = rr * 258 + n;
                float old = bf2f(baseT[idx]);
                baseT[idx] = f2bf(old + gelu(acc[nt][j] + bia));
            }
        }
    }

    // ---- bout: 256 -> 512, reduce to hmean (wave-local) + vmean (LDS + global atomics) ----
    for (int hh = 0; hh < 2; hh++) {
        for (int idx = threadIdx.x; idx < 16 * 257; idx += 128) vpart[idx] = 0.0f;
        __syncthreads();
        v8f acc[16];
#pragma unroll
        for (int nt = 0; nt < 16; nt++) acc[nt] = vzero8();
#pragma unroll
        for (int ks = 0; ks < 8; ks++) {
            v16bf a = gather_nl(rowp, ks * 32 + half * 8);
#pragma unroll
            for (int nt = 0; nt < 16; nt++) {
                v16bf bb = loadfrag(boutP + (size_t)((hh * 16 + nt) * 8 + ks) * 512 + lane * 16);
                acc[nt] = wmma_bf(a, bb, acc[nt]);
            }
        }
#pragma unroll
        for (int nt = 0; nt < 16; nt++) {
            int nl = nt * 16 + cc;
            int n  = hh * 256 + nl;
            float bia = bout_b[n];
            float colsum = 0.0f;
            float vv[8];
#pragma unroll
            for (int j = 0; j < 8; j++) { vv[j] = acc[nt][j] + bia; colsum += vv[j]; }
            colsum += __shfl_xor(colsum, 16);
            if (lane < 16)
                hmean[(size_t)(bc * 16 + pi) * 512 + n] = colsum * 0.0625f;
#pragma unroll
            for (int j = 0; j < 8; j++)
                atomicAdd(&vpart[(j + half * 8) * 257 + nl], vv[j]);
        }
        __syncthreads();
        for (int idx = threadIdx.x; idx < 4096; idx += 128) {
            int rr = idx >> 8;
            int nl = idx & 255;
            atomicAdd(&vmean[(size_t)(bc * 16 + rr) * 512 + hh * 256 + nl],
                      vpart[rr * 257 + nl] * 0.0625f);
        }
        __syncthreads();
    }
}

// ---------------- K3: tokproj (512->256->512->64, gelus) -> A4 packed (vcomp/hcomp) ----------------
// 1 wave per (side, b*c). side 0: vmean -> A4v, side 1: hmean -> A4h.
__global__ __launch_bounds__(32) void k_tokproj(
    const float* __restrict__ vmean, const float* __restrict__ hmean,
    const unsigned short* __restrict__ v1P, const unsigned short* __restrict__ v2P,
    const unsigned short* __restrict__ v3P,
    const unsigned short* __restrict__ h1P, const unsigned short* __restrict__ h2P,
    const unsigned short* __restrict__ h3P,
    const float* __restrict__ v1b, const float* __restrict__ v2b, const float* __restrict__ v3b,
    const float* __restrict__ h1b, const float* __restrict__ h2b, const float* __restrict__ h3b,
    unsigned short* __restrict__ A4v, unsigned short* __restrict__ A4h) {
    __shared__ unsigned short m0S[16 * 520];
    __shared__ unsigned short t1S[16 * 264];
    const int lane = threadIdx.x;
    const int half = lane >> 4;
    const int cc   = lane & 15;
    const int side = blockIdx.x >> 9;
    const int bc   = blockIdx.x & 511;
    const float* mean = side ? hmean : vmean;
    const unsigned short* w1 = side ? h1P : v1P;
    const unsigned short* w2 = side ? h2P : v2P;
    const unsigned short* w3 = side ? h3P : v3P;
    const float* b1 = side ? h1b : v1b;
    const float* b2 = side ? h2b : v2b;
    const float* b3 = side ? h3b : v3b;
    unsigned short* A4 = side ? A4h : A4v;

    for (int idx = lane; idx < 8192; idx += 32) {
        int rr = idx >> 9, c2 = idx & 511;
        m0S[rr * 520 + c2] = f2bf(mean[(size_t)bc * 8192 + idx]);
    }
    __syncthreads();
    const unsigned short* row0 = m0S + cc * 520;
    const unsigned short* row1 = t1S + cc * 264;

    // z1 = gelu(m0 @ w1 + b1)  [16 x 256], K=512
    {
        v8f acc[16];
#pragma unroll
        for (int nt = 0; nt < 16; nt++) acc[nt] = vzero8();
        for (int ks = 0; ks < 16; ks++) {
            v16bf a = gather_nl(row0, ks * 32 + half * 8);
#pragma unroll
            for (int nt = 0; nt < 16; nt++) {
                v16bf bb = loadfrag(w1 + (size_t)(nt * 16 + ks) * 512 + lane * 16);
                acc[nt] = wmma_bf(a, bb, acc[nt]);
            }
        }
#pragma unroll
        for (int nt = 0; nt < 16; nt++) {
            int n = nt * 16 + cc;
            float bia = b1[n];
#pragma unroll
            for (int j = 0; j < 8; j++)
                t1S[(j + half * 8) * 264 + n] = f2bf(gelu(acc[nt][j] + bia));
        }
    }
    __syncthreads();
    // z2 = gelu(z1 @ w2 + b2)  [16 x 512] -> back into m0S
    for (int hh = 0; hh < 2; hh++) {
        v8f acc[16];
#pragma unroll
        for (int nt = 0; nt < 16; nt++) acc[nt] = vzero8();
        for (int ks = 0; ks < 8; ks++) {
            v16bf a = gather_nl(row1, ks * 32 + half * 8);
#pragma unroll
            for (int nt = 0; nt < 16; nt++) {
                v16bf bb = loadfrag(w2 + (size_t)((hh * 16 + nt) * 8 + ks) * 512 + lane * 16);
                acc[nt] = wmma_bf(a, bb, acc[nt]);
            }
        }
#pragma unroll
        for (int nt = 0; nt < 16; nt++) {
            int n = hh * 256 + nt * 16 + cc;
            float bia = b2[n];
#pragma unroll
            for (int j = 0; j < 8; j++)
                m0S[(j + half * 8) * 520 + n] = f2bf(gelu(acc[nt][j] + bia));
        }
    }
    __syncthreads();
    // z3 = gelu(z2 @ w3 + b3)  [16 x 64] -> scatter into A4 packed (K4 row = bc, k = pj*64+hd)
    {
        v8f acc[4];
#pragma unroll
        for (int nt = 0; nt < 4; nt++) acc[nt] = vzero8();
        for (int ks = 0; ks < 16; ks++) {
            v16bf a = gather_nl(row0, ks * 32 + half * 8);
#pragma unroll
            for (int nt = 0; nt < 4; nt++) {
                v16bf bb = loadfrag(w3 + (size_t)(nt * 16 + ks) * 512 + lane * 16);
                acc[nt] = wmma_bf(a, bb, acc[nt]);
            }
        }
        int tile4 = bc >> 4, r4 = bc & 15;
#pragma unroll
        for (int nt = 0; nt < 4; nt++) {
            int hd = nt * 16 + cc;
            float bia = b3[hd];
#pragma unroll
            for (int j = 0; j < 8; j++) {
                int pj = j + half * 8;
                float val = gelu(acc[nt][j] + bia);
                int k = pj * 64 + hd;
                int ks4 = k >> 5, kk = k & 31;
                int h4 = (kk >> 3) & 1;
                int e4 = (kk & 7) + ((kk >> 4) << 3);
                A4[(size_t)((tile4 * 32 + ks4) * 32 + (h4 << 4) + r4) * 16 + e4] = f2bf(val);
            }
        }
    }
}

// ---------------- K4: bigproj (1024->1024 gelu, 1024->384) -> A5 packed ----------------
__global__ __launch_bounds__(32) void k_bigproj(
    const unsigned short* __restrict__ A4v, const unsigned short* __restrict__ A4h,
    const unsigned short* __restrict__ Vp1P, const unsigned short* __restrict__ Vp2P,
    const unsigned short* __restrict__ Hp1P, const unsigned short* __restrict__ Hp2P,
    const float* __restrict__ Vp1b, const float* __restrict__ Vp2b,
    const float* __restrict__ Hp1b, const float* __restrict__ Hp2b,
    unsigned short* __restrict__ A5v, unsigned short* __restrict__ A5h) {
    __shared__ unsigned short z1S[16 * 1040];
    const int lane = threadIdx.x;
    const int half = lane >> 4;
    const int cc   = lane & 15;
    const int side = blockIdx.x >> 5;
    const int tile = blockIdx.x & 31;
    const unsigned short* A4 = side ? A4h : A4v;
    const unsigned short* w1 = side ? Hp1P : Vp1P;
    const unsigned short* w2 = side ? Hp2P : Vp2P;
    const float* b1 = side ? Hp1b : Vp1b;
    const float* b2 = side ? Hp2b : Vp2b;
    unsigned short* A5 = side ? A5h : A5v;
    const unsigned short* rowz = z1S + cc * 1040;

    // z1 = gelu(A4 @ w1 + b1) [16 x 1024], K=1024
    for (int q = 0; q < 4; q++) {
        v8f acc[16];
#pragma unroll
        for (int nt = 0; nt < 16; nt++) acc[nt] = vzero8();
        for (int ks = 0; ks < 32; ks++) {
            v16bf a = loadfrag(A4 + (size_t)((tile * 32 + ks) * 32 + lane) * 16);
#pragma unroll
            for (int nt = 0; nt < 16; nt++) {
                v16bf bb = loadfrag(w1 + (size_t)((q * 16 + nt) * 32 + ks) * 512 + lane * 16);
                acc[nt] = wmma_bf(a, bb, acc[nt]);
            }
        }
#pragma unroll
        for (int nt = 0; nt < 16; nt++) {
            int n = q * 256 + nt * 16 + cc;
            float bia = b1[n];
#pragma unroll
            for (int j = 0; j < 8; j++)
                z1S[(j + half * 8) * 1040 + n] = f2bf(gelu(acc[nt][j] + bia));
        }
    }
    __syncthreads();
    // z2 = z1 @ w2 + b2  [16 x 384] -> scatter into A5 packed (row=(b*3+r)*128+n, k=c)
    for (int g = 0; g < 2; g++) {
        v8f acc[12];
#pragma unroll
        for (int nt = 0; nt < 12; nt++) acc[nt] = vzero8();
        for (int ks = 0; ks < 32; ks++) {
            v16bf a = gather_nl(rowz, ks * 32 + half * 8);
#pragma unroll
            for (int nt = 0; nt < 12; nt++) {
                v16bf bb = loadfrag(w2 + (size_t)((g * 12 + nt) * 32 + ks) * 512 + lane * 16);
                acc[nt] = wmma_bf(a, bb, acc[nt]);
            }
        }
#pragma unroll
        for (int nt = 0; nt < 12; nt++) {
            int n = (g * 12 + nt) * 16 + cc;
            float bia = b2[n];
            int rrank = n >> 7;
            int nn = n & 127;
#pragma unroll
            for (int j = 0; j < 8; j++) {
                int rr = j + half * 8;
                int rowK4 = tile * 16 + rr;
                int b = rowK4 >> 6;
                int ch = rowK4 & 63;
                int row5 = (b * 3 + rrank) * 128 + nn;
                int tile5 = row5 >> 4, r5 = row5 & 15;
                int ks5 = ch >> 5, kk = ch & 31;
                int h5 = (kk >> 3) & 1;
                int e5 = (kk & 7) + ((kk >> 4) << 3);
                A5[(size_t)((tile5 * 2 + ks5) * 32 + (h5 << 4) + r5) * 16 + e5] =
                    f2bf(acc[nt][j] + bia);
            }
        }
    }
}

// ---------------- K5: channlin (64->32 gelu, 32->64) -> Vc/Hc f32 [(b*3+r)*128+n][64] ----------------
__global__ __launch_bounds__(32) void k_channlin(
    const unsigned short* __restrict__ A5v, const unsigned short* __restrict__ A5h,
    const unsigned short* __restrict__ vc1P, const unsigned short* __restrict__ vc2P,
    const unsigned short* __restrict__ hc1P, const unsigned short* __restrict__ hc2P,
    const float* __restrict__ vc1b, const float* __restrict__ vc2b,
    const float* __restrict__ hc1b, const float* __restrict__ hc2b,
    float* __restrict__ Vc, float* __restrict__ Hc) {
    __shared__ unsigned short zS[16 * 40];
    const int lane = threadIdx.x;
    const int half = lane >> 4;
    const int cc   = lane & 15;
    const int side = (blockIdx.x >= 192) ? 1 : 0;
    const int tile = blockIdx.x - side * 192;
    const unsigned short* A5 = side ? A5h : A5v;
    const unsigned short* w1 = side ? hc1P : vc1P;
    const unsigned short* w2 = side ? hc2P : vc2P;
    const float* b1 = side ? hc1b : vc1b;
    const float* b2 = side ? hc2b : vc2b;
    float* outp = side ? Hc : Vc;

    // z = gelu(A5 @ w1 + b1) [16 x 32], K=64
    {
        v8f acc[2];
#pragma unroll
        for (int nt = 0; nt < 2; nt++) acc[nt] = vzero8();
#pragma unroll
        for (int ks = 0; ks < 2; ks++) {
            v16bf a = loadfrag(A5 + (size_t)((tile * 2 + ks) * 32 + lane) * 16);
#pragma unroll
            for (int nt = 0; nt < 2; nt++) {
                v16bf bb = loadfrag(w1 + (size_t)(nt * 2 + ks) * 512 + lane * 16);
                acc[nt] = wmma_bf(a, bb, acc[nt]);
            }
        }
#pragma unroll
        for (int nt = 0; nt < 2; nt++) {
            int n = nt * 16 + cc;
            float bia = b1[n];
#pragma unroll
            for (int j = 0; j < 8; j++)
                zS[(j + half * 8) * 40 + n] = f2bf(gelu(acc[nt][j] + bia));
        }
    }
    __syncthreads();
    // out = z @ w2 + b2 [16 x 64], K=32
    {
        const unsigned short* rowz = zS + cc * 40;
        v8f acc[4];
#pragma unroll
        for (int nt = 0; nt < 4; nt++) acc[nt] = vzero8();
        v16bf a = gather_nl(rowz, half * 8);
#pragma unroll
        for (int nt = 0; nt < 4; nt++) {
            v16bf bb = loadfrag(w2 + (size_t)nt * 512 + lane * 16);
            acc[nt] = wmma_bf(a, bb, acc[nt]);
        }
#pragma unroll
        for (int nt = 0; nt < 4; nt++) {
            int co = nt * 16 + cc;
            float bia = b2[co];
#pragma unroll
            for (int j = 0; j < 8; j++) {
                int row5 = tile * 16 + j + half * 8;
                outp[(size_t)row5 * 64 + co] = acc[nt][j] + bia;
            }
        }
    }
}

// ---------------- K6: rank-3 outer product: out[b,co,h,w] = sum_r Vc[.,h,co]*Hc[.,w,co] ----------------
__global__ __launch_bounds__(128) void k_outer(
    const float* __restrict__ Vc, const float* __restrict__ Hc, float* __restrict__ out) {
    __shared__ float HcS[128 * 65];
    __shared__ float VcS[64];
    const int b = blockIdx.x >> 7;
    const int h = blockIdx.x & 127;
    const int t = threadIdx.x;   // w
    float acc[64];
#pragma unroll
    for (int co = 0; co < 64; co++) acc[co] = 0.0f;
    for (int r = 0; r < 3; r++) {
        __syncthreads();
        for (int idx = t; idx < 8192; idx += 128) {
            int w = idx >> 6, co = idx & 63;
            HcS[w * 65 + co] = Hc[(size_t)(b * 3 + r) * 8192 + idx];
        }
        if (t < 64) VcS[t] = Vc[(size_t)((b * 3 + r) * 128 + h) * 64 + t];
        __syncthreads();
#pragma unroll
        for (int co = 0; co < 64; co++)
            acc[co] += VcS[co] * HcS[t * 65 + co];
    }
#pragma unroll
    for (int co = 0; co < 64; co++)
        out[(size_t)((b * 64 + co) * 128 + h) * 128 + t] = acc[co];
}

// ---------------- host ----------------
extern "C" void kernel_launch(void* const* d_in, const int* in_sizes, int n_in,
                              void* d_out, int out_size, void* d_ws, size_t ws_size,
                              hipStream_t stream) {
    (void)in_sizes; (void)n_in; (void)out_size; (void)ws_size;
    const float* x      = (const float*)d_in[0];
    const float* conv_w = (const float*)d_in[1];
    const float* bin_w  = (const float*)d_in[2];
    const float* bin_b  = (const float*)d_in[3];
    const float* blk_w  = (const float*)d_in[4];
    const float* blk_b  = (const float*)d_in[5];
    const float* bout_w = (const float*)d_in[6];
    const float* bout_b = (const float*)d_in[7];
    const float* v1_w = (const float*)d_in[8];  const float* v1_b = (const float*)d_in[9];
    const float* v2_w = (const float*)d_in[10]; const float* v2_b = (const float*)d_in[11];
    const float* v3_w = (const float*)d_in[12]; const float* v3_b = (const float*)d_in[13];
    const float* h1_w = (const float*)d_in[14]; const float* h1_b = (const float*)d_in[15];
    const float* h2_w = (const float*)d_in[16]; const float* h2_b = (const float*)d_in[17];
    const float* h3_w = (const float*)d_in[18]; const float* h3_b = (const float*)d_in[19];
    const float* Vp1_w = (const float*)d_in[20]; const float* Vp1_b = (const float*)d_in[21];
    const float* Vp2_w = (const float*)d_in[22]; const float* Vp2_b = (const float*)d_in[23];
    const float* Hp1_w = (const float*)d_in[24]; const float* Hp1_b = (const float*)d_in[25];
    const float* Hp2_w = (const float*)d_in[26]; const float* Hp2_b = (const float*)d_in[27];
    const float* vc1_w = (const float*)d_in[28]; const float* vc1_b = (const float*)d_in[29];
    const float* vc2_w = (const float*)d_in[30]; const float* vc2_b = (const float*)d_in[31];
    const float* hc1_w = (const float*)d_in[32]; const float* hc1_b = (const float*)d_in[33];
    const float* hc2_w = (const float*)d_in[34]; const float* hc2_b = (const float*)d_in[35];
    float* outp = (float*)d_out;

    char* ws = (char*)d_ws;
    size_t off = 0;
    auto take = [&](size_t bytes) { size_t r = off; off += (bytes + 255) & ~(size_t)255; return r; };
    size_t oP    = take(131072ull * 64 * 2);
    size_t oBin  = take(64 * 256 * 2);
    size_t oBlk  = take(4ull * 256 * 256 * 2);
    size_t oBout = take(256 * 512 * 2);
    size_t oV1 = take(512 * 256 * 2), oV2 = take(256 * 512 * 2), oV3 = take(512 * 64 * 2);
    size_t oH1 = take(512 * 256 * 2), oH2 = take(256 * 512 * 2), oH3 = take(512 * 64 * 2);
    size_t oVp1 = take(1024 * 1024 * 2), oVp2 = take(1024 * 384 * 2);
    size_t oHp1 = take(1024 * 1024 * 2), oHp2 = take(1024 * 384 * 2);
    size_t oVc1 = take(64 * 32 * 2), oVc2 = take(32 * 64 * 2);
    size_t oHc1 = take(64 * 32 * 2), oHc2 = take(32 * 64 * 2);
    size_t oVmean = take(8192ull * 512 * 4), oHmean = take(8192ull * 512 * 4);
    size_t oA4v = take(512ull * 1024 * 2), oA4h = take(512ull * 1024 * 2);
    size_t oA5v = take(3072ull * 64 * 2),  oA5h = take(3072ull * 64 * 2);
    size_t oVc = take(3072ull * 64 * 4),   oHc = take(3072ull * 64 * 4);

    auto u16 = [&](size_t o) { return (unsigned short*)(ws + o); };
    auto f32 = [&](size_t o) { return (float*)(ws + o); };

    auto pack = [&](const float* src, size_t dst, int K, int N) {
        int total = K * N;
        k_pack<<<(total + 255) / 256, 256, 0, stream>>>(src, u16(dst), K, N);
    };
    pack(bin_w, oBin, 64, 256);
    for (int i = 0; i < 4; i++)
        pack(blk_w + (size_t)i * 65536, oBlk + (size_t)i * 65536 * 2, 256, 256);
    pack(bout_w, oBout, 256, 512);
    pack(v1_w, oV1, 512, 256); pack(v2_w, oV2, 256, 512); pack(v3_w, oV3, 512, 64);
    pack(h1_w, oH1, 512, 256); pack(h2_w, oH2, 256, 512); pack(h3_w, oH3, 512, 64);
    pack(Vp1_w, oVp1, 1024, 1024); pack(Vp2_w, oVp2, 1024, 384);
    pack(Hp1_w, oHp1, 1024, 1024); pack(Hp2_w, oHp2, 1024, 384);
    pack(vc1_w, oVc1, 64, 32); pack(vc2_w, oVc2, 32, 64);
    pack(hc1_w, oHc1, 64, 32); pack(hc2_w, oHc2, 32, 64);

    k_zero<<<(4194304 + 255) / 256, 256, 0, stream>>>(f32(oVmean), 4194304);

    k_conv<<<32768, 256, 0, stream>>>(x, conv_w, u16(oP));

    k_btok<<<2048, 128, 0, stream>>>(u16(oP), u16(oBin), u16(oBlk), u16(oBout),
                                     bin_b, blk_b, bout_b, f32(oVmean), f32(oHmean));

    k_tokproj<<<1024, 32, 0, stream>>>(f32(oVmean), f32(oHmean),
                                       u16(oV1), u16(oV2), u16(oV3),
                                       u16(oH1), u16(oH2), u16(oH3),
                                       v1_b, v2_b, v3_b, h1_b, h2_b, h3_b,
                                       u16(oA4v), u16(oA4h));

    k_bigproj<<<64, 32, 0, stream>>>(u16(oA4v), u16(oA4h),
                                     u16(oVp1), u16(oVp2), u16(oHp1), u16(oHp2),
                                     Vp1_b, Vp2_b, Hp1_b, Hp2_b,
                                     u16(oA5v), u16(oA5h));

    k_channlin<<<384, 32, 0, stream>>>(u16(oA5v), u16(oA5h),
                                       u16(oVc1), u16(oVc2), u16(oHc1), u16(oHc2),
                                       vc1_b, vc2_b, hc1_b, hc2_b,
                                       f32(oVc), f32(oHc));

    k_outer<<<1024, 128, 0, stream>>>(f32(oVc), f32(oHc), outp);
}
